// S3TokenizerV2_927712936522
// MI455X (gfx1250) — matrix operand order repaired and checked by reading, not compiled, not run on hardware
//
#include <hip/hip_runtime.h>
#include <hip/hip_bf16.h>
#include <math.h>

#define B_    8
#define NMELS 128
#define TMEL  2048
#define D_    1280
#define H_    20
#define HD_   64
#define L_    6
#define DFF   5120
#define KSZ_  31
#define T1    1024
#define T2    512
#define MTOK  (B_*T2)   /* 4096 tokens */
#define K2    (D_*3)    /* 3840: conv2 im2col K */
#define QSCALE 0.35355339059327379f   /* 64^-0.25 */
#define FSQS   0.9990000128746033f

typedef __attribute__((ext_vector_type(16))) _Float16 v16h;
typedef __attribute__((ext_vector_type(8)))  float    v8f;

union HF16 { v16h v; float4 f[2]; };

__device__ __forceinline__ float gelu_f(float x) {
  return 0.5f * x * (1.0f + erff(x * 0.7071067811865475f));
}

/* CDNA5 async global->LDS copy (ASYNCcnt), bypasses VGPR staging.
 * VDST vgpr = LDS byte offset (low 32 bits of generic shared pointer). */
__device__ __forceinline__ void async_ld_b128(unsigned lds, const void* g) {
  unsigned long long ga = (unsigned long long)g;
  asm volatile("global_load_async_to_lds_b128 %0, %1, off"
               :: "v"(lds), "v"(ga) : "memory");
}

#if __has_builtin(__builtin_amdgcn_s_wait_asynccnt)
#define WAIT_ASYNC(n) __builtin_amdgcn_s_wait_asynccnt(n)
#else
#define WAIT_ASYNC(n) asm volatile("s_wait_asynccnt %0" :: "i"(n) : "memory")
#endif

/* ---------------- small prep kernels ---------------- */

__global__ void k_lens(const int* __restrict__ mel_len, int* __restrict__ len1,
                       int* __restrict__ len2, int* __restrict__ clen) {
  int b = threadIdx.x;
  if (b < B_) {
    int l1 = (mel_len[b] + 1) >> 1;
    int l2 = (l1 + 1) >> 1;
    len1[b] = l1; len2[b] = l2; clen[b] = l2;
  }
}

__global__ void k_h16(const float* __restrict__ src, _Float16* __restrict__ dst, int n) {
  int i = blockIdx.x * 256 + threadIdx.x;
  if (i < n) dst[i] = (_Float16)src[i];
}

/* f32 [K][N] -> f16 [N][K] (weights pre-transposed for WMMA B fragments) */
__global__ void k_wtrans(const float* __restrict__ src, _Float16* __restrict__ dst,
                         int K, int N) {
  int idx = blockIdx.x * 256 + threadIdx.x;
  if (idx >= K * N) return;
  int k = idx / N, n = idx % N;
  dst[(size_t)n * K + k] = (_Float16)src[idx];
}

/* conv1: mel(B,128,2048) -> x1(B,1280,1024), stride2 pad1 k3, mask by mel_len, GELU */
__global__ void k_conv1(const float* __restrict__ mel, const int* __restrict__ mel_len,
                        const float* __restrict__ w, const float* __restrict__ bias,
                        float* __restrict__ x1) {
  int idx = blockIdx.x * 256 + threadIdx.x;
  if (idx >= B_ * D_ * T1) return;
  int t  = idx % T1;
  int oc = (idx / T1) % D_;
  int b  = idx / (T1 * D_);
  int t0 = 2 * t - 1;
  int L  = mel_len[b];
  float acc = bias[oc];
  for (int ic = 0; ic < NMELS; ++ic) {
    const float* mp = mel + ((size_t)(b * NMELS + ic)) * TMEL;
    const float* wp = w + (size_t)(oc * NMELS + ic) * 3;
    if (t0 >= 0 && t0 < L) acc += wp[0] * mp[t0];
    if (t0 + 1 < L)        acc += wp[1] * mp[t0 + 1];
    if (t0 + 2 < L)        acc += wp[2] * mp[t0 + 2];
  }
  x1[((size_t)(b * D_ + oc)) * T1 + t] = gelu_f(acc);
}

/* im2col for conv2: -> f16 [MTOK][K2], masked by len1 */
__global__ void k_im2col(const float* __restrict__ x1, const int* __restrict__ len1,
                         _Float16* __restrict__ im) {
  int idx = blockIdx.x * 256 + threadIdx.x;
  if (idx >= MTOK * K2) return;
  int kc = idx % K2;
  int m  = idx / K2;
  int t  = m % T2, b = m / T2;
  int ic = kc / 3, tap = kc % 3;
  int tin = 2 * t + tap - 1;
  float v = 0.f;
  if (tin >= 0 && tin < len1[b])
    v = x1[((size_t)(b * D_ + ic)) * T1 + tin];
  im[idx] = (_Float16)v;
}

/* ---------------- WMMA tiled GEMM ----------------
 * C[M,N] = act(A[M,K] @ Bt[N,K]^T + bias) + res0 + res1
 * A, Bt f16; accum f32; 128x128x32 tiles, 8 waves, v_wmma_f32_16x16x32_f16.
 * Double-buffered LDS filled with global_load_async_to_lds_b128 (ASYNCcnt). */
#define BM 128
#define BN 128
#define BK 32

__global__ __launch_bounds__(256) void k_gemm(
    const _Float16* __restrict__ A, const _Float16* __restrict__ Bt,
    const float* __restrict__ bias, const float* __restrict__ res0,
    const float* __restrict__ res1, float* __restrict__ outF,
    _Float16* __restrict__ outH, int M, int N, int K, int doGelu) {
  __shared__ __align__(32) _Float16 sA[2][BM * BK];
  __shared__ __align__(32) _Float16 sB[2][BN * BK];
  const int tid  = threadIdx.x;
  const int lane = tid & 31, wave = tid >> 5;
  const int bm = blockIdx.y * BM, bn = blockIdx.x * BN;
  const int wm = (wave >> 2) * 64;   /* 2 wave-rows */
  const int wn = (wave & 3) * 32;    /* 4 wave-cols */
  const int fm    = lane & 15;
  const int fk8   = (lane >> 4) << 3;
  const int fk16  = (lane >> 4) << 4;
  const int rbase = (lane >> 4) << 3;

  /* per-thread 16B chunk coordinates (512 chunks per 128x32 tile, 2 each) */
  const int r0 = tid >> 2,          o0 = (tid & 3) << 3;
  const int r1 = (tid + 256) >> 2,  o1 = ((tid + 256) & 3) << 3;
  const _Float16* gA0 = &A[(size_t)(bm + r0) * K + o0];
  const _Float16* gA1 = &A[(size_t)(bm + r1) * K + o1];
  const _Float16* gB0 = &Bt[(size_t)(bn + r0) * K + o0];
  const _Float16* gB1 = &Bt[(size_t)(bn + r1) * K + o1];

  auto issue = [&](int k0, int buf) {
    async_ld_b128((unsigned)(size_t)&sA[buf][r0 * BK + o0], gA0 + k0);
    async_ld_b128((unsigned)(size_t)&sA[buf][r1 * BK + o1], gA1 + k0);
    async_ld_b128((unsigned)(size_t)&sB[buf][r0 * BK + o0], gB0 + k0);
    async_ld_b128((unsigned)(size_t)&sB[buf][r1 * BK + o1], gB1 + k0);
  };

  v8f z = {0.f,0.f,0.f,0.f,0.f,0.f,0.f,0.f};
  v8f acc[4][2];
#pragma unroll
  for (int i = 0; i < 4; ++i)
#pragma unroll
    for (int j = 0; j < 2; ++j) acc[i][j] = z;

  const int nk = K / BK;
  issue(0, 0);
  for (int it = 0; it < nk; ++it) {
    const int cur = it & 1;
    if (it + 1 < nk) {
      issue((it + 1) * BK, cur ^ 1);
      WAIT_ASYNC(4);   /* tile `it` (first 4 in-order ops) has landed */
    } else {
      WAIT_ASYNC(0);
    }
    __syncthreads();   /* all waves' async copies for tile `it` visible */

    HF16 af[4], bf[2];
#pragma unroll
    for (int i = 0; i < 4; ++i) {
      const _Float16* p = &sA[cur][(wm + i * 16 + fm) * BK];
      af[i].f[0] = *(const float4*)(p + fk8);
      af[i].f[1] = *(const float4*)(p + 16 + fk8);
    }
#pragma unroll
    for (int j = 0; j < 2; ++j) {
      const _Float16* p = &sB[cur][(wn + j * 16 + fm) * BK];
      bf[j].f[0] = *(const float4*)(p + fk8);
      bf[j].f[1] = *(const float4*)(p + 16 + fk8);
    }
#pragma unroll
    for (int i = 0; i < 4; ++i)
#pragma unroll
      for (int j = 0; j < 2; ++j)
        acc[i][j] = __builtin_amdgcn_wmma_f32_16x16x32_f16(
            false, af[i].v, false, bf[j].v, (short)0, acc[i][j], false, false);
    __syncthreads();   /* protect buf cur from next iteration's async writes */
  }

#pragma unroll
  for (int i = 0; i < 4; ++i)
#pragma unroll
    for (int j = 0; j < 2; ++j)
#pragma unroll
      for (int r = 0; r < 8; ++r) {
        int row = bm + wm + i * 16 + r + rbase;
        int col = bn + wn + j * 16 + fm;
        float v = acc[i][j][r];
        if (bias) v += bias[col];
        if (doGelu) v = gelu_f(v);
        size_t off = (size_t)row * N + col;
        if (res0) v += res0[off];
        if (res1) v += res1[off];
        if (outF) outF[off] = v;
        if (outH) outH[off] = (_Float16)v;
      }
}

/* ---------------- layernorm -> f16 ---------------- */
__global__ __launch_bounds__(256) void k_ln(const float* __restrict__ x,
                                            const float* __restrict__ w,
                                            const float* __restrict__ b,
                                            _Float16* __restrict__ out) {
  __shared__ float s1[256], s2[256];
  int row = blockIdx.x, tid = threadIdx.x;
  const float* xr = x + (size_t)row * D_;
  float a = 0.f, q = 0.f;
  for (int i = tid; i < D_; i += 256) { float v = xr[i]; a += v; q += v * v; }
  s1[tid] = a; s2[tid] = q; __syncthreads();
  for (int s = 128; s > 0; s >>= 1) {
    if (tid < s) { s1[tid] += s1[tid + s]; s2[tid] += s2[tid + s]; }
    __syncthreads();
  }
  float mean = s1[0] * (1.0f / D_);
  float var  = s2[0] * (1.0f / D_) - mean * mean;
  float inv  = rsqrtf(var + 1e-5f);
  for (int i = tid; i < D_; i += 256)
    out[(size_t)row * D_ + i] = (_Float16)((xr[i] - mean) * inv * w[i] + b[i]);
}

/* ---------------- RoPE + HD^-0.25 scale, -> f16 [b,h,t,hd] ---------------- */
__global__ void k_rope(const float* __restrict__ q32, const float* __restrict__ k32,
                       _Float16* __restrict__ qh, _Float16* __restrict__ kh) {
  int idx = blockIdx.x * 256 + threadIdx.x;
  if (idx >= MTOK * D_) return;
  int d = idx % D_;
  int m = idx / D_;
  int t = m % T2, b = m / T2;
  int h = d / HD_, hd = d % HD_;
  int fi = hd & 31;
  float freq = powf(10000.0f, -(float)fi * (1.0f / 32.0f));
  float sn, cs;
  sincosf((float)t * freq, &sn, &cs);
  size_t src = (size_t)m * D_ + d;
  size_t prt = (size_t)m * D_ + h * HD_ + ((hd < 32) ? hd + 32 : hd - 32);
  float sgn = (hd < 32) ? -1.f : 1.f;
  float qv = q32[src] * cs + sgn * q32[prt] * sn;
  float kv = k32[src] * cs + sgn * k32[prt] * sn;
  size_t dst = ((size_t)(b * H_ + h) * T2 + t) * HD_ + hd;
  qh[dst] = (_Float16)(qv * QSCALE);
  kh[dst] = (_Float16)(kv * QSCALE);
}

/* V f32 [b,t,h*hd] -> f16 transposed [b,h,hd,t] (for PV B fragments) */
__global__ void k_vprep(const float* __restrict__ v32, _Float16* __restrict__ vt) {
  int idx = blockIdx.x * 256 + threadIdx.x;
  if (idx >= MTOK * D_) return;
  int d = idx % D_;
  int m = idx / D_;
  int t = m % T2, b = m / T2;
  int h = d / HD_, hd = d % HD_;
  vt[((size_t)(b * H_ + h) * HD_ + hd) * T2 + t] = (_Float16)v32[(size_t)m * D_ + d];
}

/* FSMN: fsm = mask * (depthwise_conv31(v*mask) + v*mask) */
__global__ void k_fsmn(const float* __restrict__ v32, const float* __restrict__ w,
                       const int* __restrict__ len2, float* __restrict__ fsm) {
  int idx = blockIdx.x * 256 + threadIdx.x;
  if (idx >= MTOK * D_) return;
  int d = idx % D_;
  int m = idx / D_;
  int t = m % T2, b = m / T2;
  int lenb = len2[b];
  if (t >= lenb) { fsm[idx] = 0.f; return; }
  const float* vb = v32 + (size_t)(b * T2) * D_ + d;
  float acc = vb[(size_t)t * D_];   /* + vm term */
  const float* wd = w + d * KSZ_;
#pragma unroll
  for (int j = 0; j < KSZ_; ++j) {
    int tp = t + j - (KSZ_ - 1) / 2;
    if (tp >= 0 && tp < lenb) acc += wd[j] * vb[(size_t)tp * D_];
  }
  fsm[idx] = acc;
}

/* ---------------- flash attention, 1 wave per (b,h, 16-query tile) -------- */
__global__ __launch_bounds__(32) void k_attn(
    const _Float16* __restrict__ Q, const _Float16* __restrict__ Kh,
    const _Float16* __restrict__ Vt, const int* __restrict__ len2,
    _Float16* __restrict__ O) {
  __shared__ __align__(32) _Float16 sP[16 * 32];
  const int bh = blockIdx.x, qt = blockIdx.y;
  const int b = bh / H_, h = bh % H_;
  const int lane  = threadIdx.x;
  const int fm    = lane & 15;
  const int fk8   = (lane >> 4) << 3;
  const int fk16  = (lane >> 4) << 4;
  const int rbase = (lane >> 4) << 3;
  const int lenb  = len2[b];
  const _Float16* Qb = Q  + (size_t)bh * T2 * HD_;
  const _Float16* Kb = Kh + (size_t)bh * T2 * HD_;
  const _Float16* Vb = Vt + (size_t)bh * HD_ * T2;

  HF16 qf[2];
#pragma unroll
  for (int c = 0; c < 2; ++c) {
    const _Float16* p = &Qb[(qt * 16 + fm) * HD_ + c * 32];
    qf[c].f[0] = *(const float4*)(p + fk8);
    qf[c].f[1] = *(const float4*)(p + 16 + fk8);
  }
  v8f z = {0.f,0.f,0.f,0.f,0.f,0.f,0.f,0.f};
  float mrow[8], lrow[8];
  v8f oacc[4];
#pragma unroll
  for (int r = 0; r < 8; ++r) { mrow[r] = -3.0e38f; lrow[r] = 0.f; }
#pragma unroll
  for (int j = 0; j < 4; ++j) oacc[j] = z;

  for (int kt = 0; kt < T2 / 32; ++kt) {
    v8f s0 = z, s1v = z;
#pragma unroll
    for (int sub = 0; sub < 2; ++sub) {
      int key0 = kt * 32 + sub * 16;
      v8f s = z;
#pragma unroll
      for (int c = 0; c < 2; ++c) {
        HF16 kf;
        const _Float16* p = &Kb[(key0 + fm) * HD_ + c * 32];
        kf.f[0] = *(const float4*)(p + fk8);
        kf.f[1] = *(const float4*)(p + 16 + fk8);
        s = __builtin_amdgcn_wmma_f32_16x16x32_f16(
            false, qf[c].v, false, kf.v, (short)0, s, false, false);
      }
      if (sub == 0) s0 = s; else s1v = s;
    }
    int kc0 = kt * 32 + fm;
    float b0 = (kc0 < lenb)      ? 0.f : -1.0e10f;
    float b1 = (kc0 + 16 < lenb) ? 0.f : -1.0e10f;
#pragma unroll
    for (int r = 0; r < 8; ++r) {
      float v0 = s0[r] + b0, v1 = s1v[r] + b1;
      float mx = fmaxf(v0, v1);
      mx = fmaxf(mx, __shfl_xor(mx, 1));
      mx = fmaxf(mx, __shfl_xor(mx, 2));
      mx = fmaxf(mx, __shfl_xor(mx, 4));
      mx = fmaxf(mx, __shfl_xor(mx, 8));
      float mnew  = fmaxf(mrow[r], mx);
      float alpha = expf(mrow[r] - mnew);
      float p0 = expf(v0 - mnew);
      float p1 = expf(v1 - mnew);
      float ps = p0 + p1;
      ps += __shfl_xor(ps, 1);
      ps += __shfl_xor(ps, 2);
      ps += __shfl_xor(ps, 4);
      ps += __shfl_xor(ps, 8);
      lrow[r] = lrow[r] * alpha + ps;
      mrow[r] = mnew;
#pragma unroll
      for (int j = 0; j < 4; ++j) oacc[j][r] *= alpha;
      int prow = r + rbase;
      sP[prow * 32 + fm]      = (_Float16)p0;
      sP[prow * 32 + 16 + fm] = (_Float16)p1;
    }
    __syncthreads();
    HF16 pf;
    {
      const _Float16* p = &sP[fm * 32];
      pf.f[0] = *(const float4*)(p + fk8);
      pf.f[1] = *(const float4*)(p + 16 + fk8);
    }
#pragma unroll
    for (int j = 0; j < 4; ++j) {
      HF16 vf;
      const _Float16* p = &Vb[(j * 16 + fm) * T2 + kt * 32];
      vf.f[0] = *(const float4*)(p + fk8);
      vf.f[1] = *(const float4*)(p + 16 + fk8);
      oacc[j] = __builtin_amdgcn_wmma_f32_16x16x32_f16(
          false, pf.v, false, vf.v, (short)0, oacc[j], false, false);
    }
    __syncthreads();
  }
#pragma unroll
  for (int r = 0; r < 8; ++r) {
    float rl = 1.0f / lrow[r];
    int trow = qt * 16 + r + rbase;
#pragma unroll
    for (int j = 0; j < 4; ++j) {
      int d = h * HD_ + j * 16 + fm;
      O[((size_t)(b * T2 + trow)) * D_ + d] = (_Float16)(oacc[j][r] * rl);
    }
  }
}

/* ---------------- final projection + FSQ ---------------- */
__global__ void k_proj(const float* __restrict__ x, const float* __restrict__ pw,
                       const float* __restrict__ pb, float* __restrict__ h8) {
  int idx = blockIdx.x * 256 + threadIdx.x;
  if (idx >= MTOK * 8) return;
  int j = idx & 7, m = idx >> 3;
  const float* xr = x + (size_t)m * D_;
  float s = pb[j];
  for (int k = 0; k < D_; ++k) s += xr[k] * pw[k * 8 + j];
  h8[idx] = tanhf(s);
}

__global__ void k_code(const float* __restrict__ h8, int* __restrict__ code) {
  int m = blockIdx.x * 256 + threadIdx.x;
  if (m >= MTOK) return;
  float s = 0.f, p3 = 1.f;
  for (int j = 0; j < 8; ++j) {
    float hv = rintf(h8[m * 8 + j] * FSQS) + 1.0f;
    s += hv * p3;
    p3 *= 3.f;
  }
  code[m] = (int)s;
}

/* ---------------- host orchestration ---------------- */
extern "C" void kernel_launch(void* const* d_in, const int* in_sizes, int n_in,
                              void* d_out, int out_size, void* d_ws, size_t ws_size,
                              hipStream_t stream) {
  (void)in_sizes; (void)n_in; (void)out_size; (void)ws_size;
  const float* mel   = (const float*)d_in[0];
  const int*   mlen  = (const int*)d_in[1];
  const float* c1w   = (const float*)d_in[2];
  const float* c1b   = (const float*)d_in[3];
  const float* c2w   = (const float*)d_in[4];
  const float* c2b   = (const float*)d_in[5];
  const float* qW    = (const float*)d_in[6];
  const float* kW    = (const float*)d_in[7];
  const float* vW    = (const float*)d_in[8];
  const float* oW    = (const float*)d_in[9];
  const float* m1W   = (const float*)d_in[10];
  const float* m2W   = (const float*)d_in[11];
  const float* fW    = (const float*)d_in[12];
  const float* qB    = (const float*)d_in[13];
  const float* vB    = (const float*)d_in[14];
  const float* oB    = (const float*)d_in[15];
  const float* lnaB  = (const float*)d_in[16];
  const float* lnmB  = (const float*)d_in[17];
  const float* m2B   = (const float*)d_in[18];
  const float* m1B   = (const float*)d_in[19];
  const float* lnaW  = (const float*)d_in[20];
  const float* lnmW  = (const float*)d_in[21];
  const float* pW    = (const float*)d_in[22];
  const float* pB    = (const float*)d_in[23];

  int* outCode = (int*)d_out;
  int* outLen  = outCode + MTOK;

  char* p = (char*)d_ws;
  auto alloc = [&](size_t bytes) -> void* {
    void* r = (void*)p;
    p += (bytes + 255) & ~(size_t)255;
    return r;
  };
  int*       len1  = (int*)alloc(B_ * 4);
  int*       len2  = (int*)alloc(B_ * 4);
  float*     x     = (float*)alloc((size_t)MTOK * D_ * 4);
  _Float16*  xn16  = (_Float16*)alloc((size_t)MTOK * D_ * 2);
  float*     x1    = (float*)alloc((size_t)B_ * D_ * T1 * 4);
  _Float16*  imc   = (_Float16*)alloc((size_t)MTOK * K2 * 2);
  _Float16*  wtc2  = (_Float16*)alloc((size_t)D_ * K2 * 2);
  _Float16*  wtq   = (_Float16*)alloc((size_t)D_ * D_ * 2);
  _Float16*  wtk   = (_Float16*)alloc((size_t)D_ * D_ * 2);
  _Float16*  wtv   = (_Float16*)alloc((size_t)D_ * D_ * 2);
  _Float16*  wto   = (_Float16*)alloc((size_t)D_ * D_ * 2);
  _Float16*  wtm1  = (_Float16*)alloc((size_t)DFF * D_ * 2);
  _Float16*  wtm2  = (_Float16*)alloc((size_t)D_ * DFF * 2);
  float*     q32   = (float*)alloc((size_t)MTOK * D_ * 4);
  float*     k32   = (float*)alloc((size_t)MTOK * D_ * 4);
  float*     v32   = (float*)alloc((size_t)MTOK * D_ * 4);
  _Float16*  qh    = (_Float16*)alloc((size_t)MTOK * D_ * 2);
  _Float16*  kh    = (_Float16*)alloc((size_t)MTOK * D_ * 2);
  _Float16*  vt    = (_Float16*)alloc((size_t)MTOK * D_ * 2);
  float*     fsm   = (float*)alloc((size_t)MTOK * D_ * 4);
  _Float16*  wv16  = (_Float16*)alloc((size_t)MTOK * D_ * 2);
  _Float16*  h16   = (_Float16*)alloc((size_t)MTOK * DFF * 2);
  float*     h8    = (float*)alloc((size_t)MTOK * 8 * 4);

  const int TPB = 256;
  auto g1 = [](long n) { return (unsigned)((n + 255) / 256); };

  k_lens<<<1, B_, 0, stream>>>(mlen, len1, len2, outLen);
  /* conv2 weight (D,D,3) is already [N][K] for im2col K ordering */
  k_h16<<<g1((long)D_ * K2), TPB, 0, stream>>>(c2w, wtc2, D_ * K2);
  k_conv1<<<g1((long)B_ * D_ * T1), TPB, 0, stream>>>(mel, mlen, c1w, c1b, x1);
  k_im2col<<<g1((long)MTOK * K2), TPB, 0, stream>>>(x1, len1, imc);
  k_gemm<<<dim3(D_ / BN, MTOK / BM), TPB, 0, stream>>>(
      imc, wtc2, c2b, nullptr, nullptr, x, nullptr, MTOK, D_, K2, 1);

  for (int l = 0; l < L_; ++l) {
    k_wtrans<<<g1((long)D_ * D_), TPB, 0, stream>>>(qW + (size_t)l * D_ * D_, wtq, D_, D_);
    k_wtrans<<<g1((long)D_ * D_), TPB, 0, stream>>>(kW + (size_t)l * D_ * D_, wtk, D_, D_);
    k_wtrans<<<g1((long)D_ * D_), TPB, 0, stream>>>(vW + (size_t)l * D_ * D_, wtv, D_, D_);
    k_wtrans<<<g1((long)D_ * D_), TPB, 0, stream>>>(oW + (size_t)l * D_ * D_, wto, D_, D_);
    k_wtrans<<<g1((long)D_ * DFF), TPB, 0, stream>>>(m1W + (size_t)l * D_ * DFF, wtm1, D_, DFF);
    k_wtrans<<<g1((long)DFF * D_), TPB, 0, stream>>>(m2W + (size_t)l * DFF * D_, wtm2, DFF, D_);

    k_ln<<<MTOK, TPB, 0, stream>>>(x, lnaW + l * D_, lnaB + l * D_, xn16);
    k_gemm<<<dim3(D_ / BN, MTOK / BM), TPB, 0, stream>>>(
        xn16, wtq, qB + l * D_, nullptr, nullptr, q32, nullptr, MTOK, D_, D_, 0);
    k_gemm<<<dim3(D_ / BN, MTOK / BM), TPB, 0, stream>>>(
        xn16, wtk, nullptr, nullptr, nullptr, k32, nullptr, MTOK, D_, D_, 0);
    k_gemm<<<dim3(D_ / BN, MTOK / BM), TPB, 0, stream>>>(
        xn16, wtv, vB + l * D_, nullptr, nullptr, v32, nullptr, MTOK, D_, D_, 0);

    k_rope<<<g1((long)MTOK * D_), TPB, 0, stream>>>(q32, k32, qh, kh);
    k_vprep<<<g1((long)MTOK * D_), TPB, 0, stream>>>(v32, vt);
    k_fsmn<<<g1((long)MTOK * D_), TPB, 0, stream>>>(v32, fW + (size_t)l * D_ * KSZ_, len2, fsm);

    k_attn<<<dim3(B_ * H_, T2 / 16), 32, 0, stream>>>(qh, kh, vt, len2, wv16);

    k_gemm<<<dim3(D_ / BN, MTOK / BM), TPB, 0, stream>>>(
        wv16, wto, oB + l * D_, x, fsm, x, nullptr, MTOK, D_, D_, 0);

    k_ln<<<MTOK, TPB, 0, stream>>>(x, lnmW + l * D_, lnmB + l * D_, xn16);
    k_gemm<<<dim3(DFF / BN, MTOK / BM), TPB, 0, stream>>>(
        xn16, wtm1, m1B + (size_t)l * DFF, nullptr, nullptr, nullptr, h16, MTOK, DFF, D_, 1);
    k_gemm<<<dim3(D_ / BN, MTOK / BM), TPB, 0, stream>>>(
        h16, wtm2, m2B + l * D_, x, nullptr, x, nullptr, MTOK, D_, DFF, 0);
  }

  k_proj<<<g1((long)MTOK * 8), TPB, 0, stream>>>(x, pW, pB, h8);
  k_code<<<g1((long)MTOK), TPB, 0, stream>>>(h8, outCode);
}
